// HNSWClassifier_34059090657996
// MI455X (gfx1250) — compile-verified, matrix operand order checked by reading, not run on hardware
//
#include <hip/hip_runtime.h>
#include <hip/hip_bf16.h>
#include <stdint.h>

// Problem constants (from reference)
#define B_DIM       1024
#define IN_DIM      2048
#define FEAT_DIM    512
#define SAMPLER_NUM 8192

#define LDSTRIDE 40   // ushorts per LDS tile row: 32 data + 8 pad (80 B, kills bank conflicts)

typedef __attribute__((ext_vector_type(16))) __bf16        v16bf;
typedef __attribute__((ext_vector_type(8)))  float         v8f;
typedef __attribute__((ext_vector_type(4)))  unsigned int  v4u;
typedef __attribute__((ext_vector_type(8)))  int           v8i_t;
typedef __attribute__((ext_vector_type(4)))  int           v4i_t;

union Frag {
  uint4 u4[2];   // 32 bytes = 16 bf16
  v16bf v;
};

__device__ __forceinline__ unsigned short f2bf(float f) {
  unsigned u = __float_as_uint(f);
  unsigned r = u + 0x7FFFu + ((u >> 16) & 1u);
  return (unsigned short)(r >> 16);
}

__device__ __forceinline__ bool keep_mask(unsigned idx) {
  unsigned h = idx * 2654435761u;
  h ^= h >> 16; h *= 0x85EBCA6Bu;
  h ^= h >> 13; h *= 0xC2B2AE35u;
  h ^= h >> 16;
  return (h & 1u) != 0u;
}

#if defined(__has_builtin)
#if __has_builtin(__builtin_amdgcn_tensor_load_to_lds)
#define HAS_TDM 1
#endif
#endif

#ifdef HAS_TDM
// ---------------------------------------------------------------------------
// Issue one TDM 2D tile load: 128 rows x 32 bf16 from row-major (ld = row
// stride in elements) into LDS at lds_off, padding each 64B row with 16B
// (pad_interval code 3 = 16 DWORDs, pad_amount code 3 = 4 DWORDs).
// D# bit packing per CDNA5 ISA 8.3/8.4 (group0 128b, group1 256b).
// 6-arg builtin form: (v4u g0, v8i g1, v4i g2, v4i g3, v8i extra, i32 cpol)
// ---------------------------------------------------------------------------
__device__ __forceinline__ void tdm_load_tile(const unsigned short* gsrc, int ld,
                                              unsigned lds_off) {
  uint64_t ga = (uint64_t)(uintptr_t)gsrc;
  v4u g0;
  g0[0] = 1u;                                            // count=1, user mode
  g0[1] = lds_off;                                       // lds_addr (bytes)
  g0[2] = (unsigned)ga;                                  // global_addr[31:0]
  g0[3] = (unsigned)((ga >> 32) & 0x01FFFFFFu)           // global_addr[56:32]
          | (2u << 30);                                  // type = 2 ("image")
  unsigned dim0 = (unsigned)ld;                          // tensor width (elems)
  unsigned dim1 = 128u;                                  // rows available
  unsigned s0   = (unsigned)ld;                          // tensor_dim0_stride
  v8i_t g1;
  g1[0] = (int)((1u << 16)                               // data_size = 2 bytes
                | (1u << 20)                             // pad_enable
                | (3u << 22)                             // pad_interval: 16 DW
                | (3u << 25));                           // pad_amount: 4 DW
  g1[1] = (int)((dim0 & 0xFFFFu) << 16);                 // tensor_dim0[15:0]
  g1[2] = (int)(((dim0 >> 16) & 0xFFFFu)                 // tensor_dim0[31:16]
                | ((dim1 & 0xFFFFu) << 16));             // tensor_dim1[15:0]
  g1[3] = (int)(((dim1 >> 16) & 0xFFFFu)                 // tensor_dim1[31:16]
                | (32u << 16));                          // tile_dim0 = 32
  g1[4] = (int)128u;                                     // tile_dim1=128, tile_dim2=0
  g1[5] = (int)s0;                                       // dim0_stride[31:0]
  g1[6] = 0;                                             // stride[47:32], dim1_stride lo
  g1[7] = 0;
  v4i_t z4 = {0, 0, 0, 0};                               // 2D: groups 2/3 unused
  v8i_t z8 = {0, 0, 0, 0, 0, 0, 0, 0};
  __builtin_amdgcn_tensor_load_to_lds(g0, g1, z4, z4, z8, 0);
}
#endif

// ---------------------------------------------------------------------------
// fp32 -> bf16 streaming convert (n divisible by 4)
// ---------------------------------------------------------------------------
__global__ __launch_bounds__(256)
void cvt_f32_bf16(const float* __restrict__ in, unsigned short* __restrict__ out, int n) {
  int i = (blockIdx.x * blockDim.x + threadIdx.x) * 4;
  if (i < n) {
    float4 f = *(const float4*)(in + i);
    ushort4 o;
    o.x = f2bf(f.x); o.y = f2bf(f.y); o.z = f2bf(f.z); o.w = f2bf(f.w);
    *(ushort4*)(out + i) = o;
  }
}

// ---------------------------------------------------------------------------
// gather W[cand_idx] -> bf16 reduced weights, gather bias -> bred
// ---------------------------------------------------------------------------
__global__ __launch_bounds__(128)
void gather_w(const float* __restrict__ W, const float* __restrict__ bias,
              const int* __restrict__ cand,
              unsigned short* __restrict__ wred, float* __restrict__ bred) {
  const int row = blockIdx.x;
  const int idx = cand[row];
  const float* src = W + (size_t)idx * FEAT_DIM;
  unsigned short* dst = wred + (size_t)row * FEAT_DIM;
  for (int c = threadIdx.x * 4; c < FEAT_DIM; c += blockDim.x * 4) {
    float4 f = *(const float4*)(src + c);
    ushort4 o;
    o.x = f2bf(f.x); o.y = f2bf(f.y); o.z = f2bf(f.z); o.w = f2bf(f.w);
    *(ushort4*)(dst + c) = o;
  }
  if (threadIdx.x == 0) bred[row] = bias[idx];
}

// ---------------------------------------------------------------------------
// new_labels = searchsorted(cand_idx, cand_idx[labels]) ; written as float
// ---------------------------------------------------------------------------
__global__ __launch_bounds__(256)
void remap_labels(const int* __restrict__ labels, const int* __restrict__ cand,
                  float* __restrict__ out) {
  int i = blockIdx.x * blockDim.x + threadIdx.x;
  if (i < B_DIM) {
    int gl = cand[labels[i]];
    int lo = 0, hi = SAMPLER_NUM;
    while (lo < hi) {
      int mid = (lo + hi) >> 1;
      if (cand[mid] < gl) lo = mid + 1; else hi = mid;
    }
    out[(size_t)B_DIM * SAMPLER_NUM + i] = (float)lo;
  }
}

// ---------------------------------------------------------------------------
// bf16 WMMA GEMM with TDM -> LDS double-buffered pipeline.
// C[M,N] = A[M,K] * Brows[N,K]^T (+ bias[n]).
// Block tile 128x128, 8 waves (2x4), wave tile 64x32 = 4x2 WMMA accumulators.
// EPI=0: float C (logits).  EPI=1: dropout + bf16 store (feat).
// ---------------------------------------------------------------------------
template <int EPI>
__global__ __launch_bounds__(256)
void gemm_bf16_wmma(const unsigned short* __restrict__ A,
                    const unsigned short* __restrict__ Bm,
                    const float* __restrict__ bias,
                    void* __restrict__ Cout,
                    int M, int N, int K) {
  __shared__ unsigned short smem[2][2][128 * LDSTRIDE];   // [buf][A/B] 40 KB

  const int tid  = threadIdx.x;
  const int lane = tid & 31;
  const int wave = tid >> 5;
  const int wm   = wave >> 2;          // 0..1 (M dir)
  const int wn   = wave & 3;           // 0..3 (N dir)
  const int r    = lane & 15;
  const int half = lane >> 4;

  const int m_blk = blockIdx.y * 128;
  const int n_blk = blockIdx.x * 128;

  const unsigned short* Ag = A  + (size_t)m_blk * K;   // tile-block row start
  const unsigned short* Bg = Bm + (size_t)n_blk * K;

  v8f zero = {0.f, 0.f, 0.f, 0.f, 0.f, 0.f, 0.f, 0.f};
  v8f acc[4][2];
#pragma unroll
  for (int mi = 0; mi < 4; ++mi)
#pragma unroll
    for (int ni = 0; ni < 2; ++ni) acc[mi][ni] = zero;

  const int nk = K >> 5;   // K / 32

#ifdef HAS_TDM
  // ---- prologue: wave 0 issues TDM loads of k-slice 0 into buffer 0 ----
  if (wave == 0) {
    tdm_load_tile(Ag, K, (unsigned)(uintptr_t)&smem[0][0][0]);
    tdm_load_tile(Bg, K, (unsigned)(uintptr_t)&smem[0][1][0]);
  }
  for (int ki = 0; ki < nk; ++ki) {
    const int cur = ki & 1;
    if (wave == 0) {
      if (ki + 1 < nk) {
        // issue next slice into the other buffer, then wait for current pair
        tdm_load_tile(Ag + (size_t)(ki + 1) * 32, K,
                      (unsigned)(uintptr_t)&smem[1 - cur][0][0]);
        tdm_load_tile(Bg + (size_t)(ki + 1) * 32, K,
                      (unsigned)(uintptr_t)&smem[1 - cur][1][0]);
        __builtin_amdgcn_s_wait_tensorcnt(2);
      } else {
        __builtin_amdgcn_s_wait_tensorcnt(0);
      }
    }
    __syncthreads();                    // current buffer visible to all waves
#else
  for (int ki = 0; ki < nk; ++ki) {
    const int cur = 0;
    {
      // cooperative fallback copy: 256 threads, 128 rows x 2 16B chunks/tile
      const int crow = tid >> 1;
      const int cc   = (tid & 1) * 16;
      const size_t kof = (size_t)ki * 32;
      *(uint4*)&smem[0][0][crow * LDSTRIDE + cc] =
          *(const uint4*)(Ag + (size_t)crow * K + kof + cc);
      *(uint4*)&smem[0][1][crow * LDSTRIDE + cc] =
          *(const uint4*)(Bg + (size_t)crow * K + kof + cc);
    }
    __syncthreads();
#endif

    // ---- compute: per-wave fragment reads from LDS (ds_load_b128) ----
    Frag fa[4], fb[2];
#pragma unroll
    for (int mi = 0; mi < 4; ++mi) {
      const unsigned short* p =
          &smem[cur][0][(wm * 64 + mi * 16 + r) * LDSTRIDE + half * 8];
      fa[mi].u4[0] = *(const uint4*)(p);        // K: half*8 .. half*8+7
      fa[mi].u4[1] = *(const uint4*)(p + 16);   // K: 16+half*8 ..
    }
#pragma unroll
    for (int ni = 0; ni < 2; ++ni) {
      const unsigned short* p =
          &smem[cur][1][(wn * 32 + ni * 16 + r) * LDSTRIDE + half * 8];
      fb[ni].u4[0] = *(const uint4*)(p);
      fb[ni].u4[1] = *(const uint4*)(p + 16);
    }
#pragma unroll
    for (int mi = 0; mi < 4; ++mi)
#pragma unroll
      for (int ni = 0; ni < 2; ++ni)
        acc[mi][ni] = __builtin_amdgcn_wmma_f32_16x16x32_bf16(
            false, fa[mi].v, false, fb[ni].v, (short)0, acc[mi][ni],
            false, false);

    __syncthreads();                    // all reads done before buffer reuse
  }

  // ---- epilogue: lane(0..15)=col, half selects M+8; VGPR v = row ----
#pragma unroll
  for (int mi = 0; mi < 4; ++mi) {
#pragma unroll
    for (int ni = 0; ni < 2; ++ni) {
      const int n  = n_blk + wn * 32 + ni * 16 + r;
      const int m0 = m_blk + wm * 64 + mi * 16 + half * 8;
      const float bv = bias[n];
#pragma unroll
      for (int v = 0; v < 8; ++v) {
        float val = acc[mi][ni][v] + bv;
        const size_t idx = (size_t)(m0 + v) * N + n;
        if (EPI == 0) {
          ((float*)Cout)[idx] = val;
        } else {
          val = keep_mask((unsigned)idx) ? val * 2.0f : 0.0f;  // p=0.5 -> x2
          ((unsigned short*)Cout)[idx] = f2bf(val);
        }
      }
    }
  }
}

// ---------------------------------------------------------------------------
extern "C" void kernel_launch(void* const* d_in, const int* in_sizes, int n_in,
                              void* d_out, int out_size, void* d_ws, size_t ws_size,
                              hipStream_t stream) {
  const float* x      = (const float*)d_in[0];
  const int*   labels = (const int*)d_in[1];
  const int*   cand   = (const int*)d_in[2];
  const float* base_w = (const float*)d_in[3];
  const float* base_b = (const float*)d_in[4];
  const float* W      = (const float*)d_in[5];
  const float* bias   = (const float*)d_in[6];

  unsigned short* x_bf    = (unsigned short*)d_ws;
  unsigned short* bw_bf   = x_bf    + (size_t)B_DIM * IN_DIM;
  unsigned short* feat_bf = bw_bf   + (size_t)FEAT_DIM * IN_DIM;
  unsigned short* wred    = feat_bf + (size_t)B_DIM * FEAT_DIM;
  float*          bred    = (float*)(wred + (size_t)SAMPLER_NUM * FEAT_DIM);

  cvt_f32_bf16<<<(B_DIM * IN_DIM / 4 + 255) / 256, 256, 0, stream>>>(
      x, x_bf, B_DIM * IN_DIM);
  cvt_f32_bf16<<<(FEAT_DIM * IN_DIM / 4 + 255) / 256, 256, 0, stream>>>(
      base_w, bw_bf, FEAT_DIM * IN_DIM);
  gather_w<<<SAMPLER_NUM, 128, 0, stream>>>(W, bias, cand, wred, bred);
  remap_labels<<<(B_DIM + 255) / 256, 256, 0, stream>>>(labels, cand, (float*)d_out);

  // GEMM1: feat = dropout(x @ base_w^T + base_b)  [1024 x 512], bf16 out
  gemm_bf16_wmma<1><<<dim3(FEAT_DIM / 128, B_DIM / 128), 256, 0, stream>>>(
      x_bf, bw_bf, base_b, feat_bf, B_DIM, FEAT_DIM, IN_DIM);

  // GEMM2: logits = feat @ wred^T + bred  [1024 x 8192], fp32 out
  gemm_bf16_wmma<0><<<dim3(SAMPLER_NUM / 128, B_DIM / 128), 256, 0, stream>>>(
      feat_bf, wred, bred, d_out, B_DIM, SAMPLER_NUM, FEAT_DIM);
}
